// FASTMultiHeadAttention_67980742361468
// MI455X (gfx1250) — matrix operand order repaired and checked by reading, not compiled, not run on hardware
//
#include <hip/hip_runtime.h>
#include <hip/hip_bf16.h>

// fastmax (2nd-order Taylor softmax attention) forward, causal, B=1 H=8 N=2048 D=64.
// Pre-pass: fp32 -> bf16 repack of K (row-major), V (transposed), RPE into d_ws.
// Hot loop: v_wmma_f32_16x16x32_bf16 everywhere + double-buffered async K->LDS copies.

typedef __attribute__((ext_vector_type(16))) __bf16 v16bf;
typedef __attribute__((ext_vector_type(8)))  float  v8f;
typedef __attribute__((ext_vector_type(4)))  int    v4i;

#define NN 2048
#define DD 64
#define HH 8

#if __has_builtin(__builtin_amdgcn_global_load_async_to_lds_b128)
#define HAVE_ASYNC_LDS 1
#else
#define HAVE_ASYNC_LDS 0
#endif

#if __has_builtin(__builtin_amdgcn_s_wait_asynccnt)
#define WAIT_ASYNC(n) __builtin_amdgcn_s_wait_asynccnt(n)
#else
#define WAIT_ASYNC(n) asm volatile("s_wait_asynccnt %0" ::"i"(n) : "memory")
#endif

__device__ __forceinline__ __bf16 tobf(float f) { return (__bf16)f; }

// 16-bit A-matrix 16x32 layout: lane half hf, vector element e -> K index.
// Per lane this is one contiguous 16-wide run: {16hf..16hf+15} (matches B loads).
__device__ __forceinline__ int a_kmap(int e, int hf) {
    return 16 * hf + e;
}

#define WMMA(A, B, C) \
    __builtin_amdgcn_wmma_f32_16x16x32_bf16(false, (A), false, (B), (short)0, (C), false, false)

#if HAVE_ASYNC_LDS
// builtin signature (per hipcc diagnostic): param0 = AS1 v4i* (global), then LDS ptr,
// imm offset, imm cpol.
typedef __attribute__((address_space(1))) v4i* gptr_v4i;
typedef __attribute__((address_space(3))) v4i* lptr_v4i;

__device__ __forceinline__ void async_b128_to_lds(const __bf16* gsrc, __bf16* ldst) {
    __builtin_amdgcn_global_load_async_to_lds_b128(
        (gptr_v4i)(void*)gsrc,
        (lptr_v4i)(void*)ldst,
        0, 0);
}
#endif

// ---------------- pre-pass kernels ----------------

__global__ __launch_bounds__(256) void cvt_bf16(const float* __restrict__ src,
                                                __bf16* __restrict__ dst, int n) {
    int i = blockIdx.x * 256 + threadIdx.x;
    if (i < n) dst[i] = tobf(src[i]);
}

// vt[h][d][j] = v[h][j][d]   (per-head transpose, bf16)
__global__ __launch_bounds__(256) void transpose_v_bf16(const float* __restrict__ v,
                                                        __bf16* __restrict__ vt) {
    int idx = blockIdx.x * 256 + threadIdx.x;   // over H*N*D
    int d = idx % DD;
    int j = (idx / DD) % NN;
    int h = idx / (DD * NN);
    vt[((size_t)h * DD + d) * NN + j] = tobf(v[idx]);
}

// ---------------- main kernel ----------------

__global__ __launch_bounds__(128)
void fastmax_fwd(const float*  __restrict__ q,
                 const __bf16* __restrict__ kb,    // [H][N][D] bf16
                 const __bf16* __restrict__ vtb,   // [H][D][N] bf16
                 const __bf16* __restrict__ rb,    // [2N-1][D] bf16
                 float* __restrict__ out) {
    __shared__ float lds_qr[4][16][48];  // per-wave qr band tile (f32)
    __shared__ float lds_w [4][16][32];  // per-wave weight tile pair (f32)
#if HAVE_ASYNC_LDS
    __shared__ __bf16 lds_kst[4][2][32 * DD];  // per-wave double-buffered K tile pair
#endif

    const int lane = threadIdx.x & 31;
    const int wid  = threadIdx.x >> 5;
    const int m16  = lane & 15;   // column / within-half index
    const int hf   = lane >> 4;   // lane half

    const int h  = blockIdx.x / (NN / 64);
    const int rb_idx = blockIdx.x % (NN / 64);
    const int i0 = rb_idx * 64 + wid * 16;   // this wave's 16-row tile
    const int it = i0 >> 4;                  // diagonal column-tile index

    const size_t hq = (size_t)h * NN * DD;

#if HAVE_ASYNC_LDS
    // async copy of the K rows for column-tile pair jtp into per-wave buffer `buf`:
    // 32 rows x 128B = 256 chunks of 16B; 8 chunks (=8 async instructions) per lane.
    auto issue_kpair = [&](int jtp, int buf) {
        #pragma unroll
        for (int ci = 0; ci < 8; ++ci) {
            int chunk = ci * 32 + lane;           // 0..255
            int rr    = chunk >> 3;               // row within pair, 0..31
            int col   = (chunk & 7) * 8;          // element column, step 8 bf16 = 16B
            int j = 16 * jtp + rr;  if (j > NN - 1) j = NN - 1;
            async_b128_to_lds(kb + hq + (size_t)j * DD + col,
                              &lds_kst[wid][buf][chunk * 8]);
        }
    };
#endif

    // ---- Q A-fragments: 16x64 bf16 as two K=32 fragments (one-time, f32 gather) ----
    const float* qrow = q + hq + (size_t)(i0 + m16) * DD;
    v16bf aq0, aq1;
    #pragma unroll
    for (int e = 0; e < 16; ++e) {
        int kk = a_kmap(e, hf);
        aq0[e] = tobf(qrow[kk]);
        aq1[e] = tobf(qrow[kk + 32]);
    }

    v8f acc_o[4];
    acc_o[0] = (v8f){}; acc_o[1] = (v8f){}; acc_o[2] = (v8f){}; acc_o[3] = (v8f){};
    float rs_part[8] = {0.f, 0.f, 0.f, 0.f, 0.f, 0.f, 0.f, 0.f};

    v8f qr_hi = (v8f){};   // carries cols 32..47 of the rpe window (reuse across iters)

#if HAVE_ASYNC_LDS
    issue_kpair(0, 0);     // prologue: stage first K tile pair
#endif
    int cur = 0;

    for (int jt = 0; jt <= it; jt += 2) {
        const int j0pair = jt << 4;
        const int m0 = i0 - j0pair + NN - 32;  // rpe window: cols c=0..47 <-> m = m0+c

#if HAVE_ASYNC_LDS
        const bool pre = (jt + 2 <= it);       // wave-uniform
        if (pre) issue_kpair(jt + 2, cur ^ 1); // overlap copy with the qr WMMAs below
#endif

        // ---- qr band via WMMA: qr[i, c] = q_i . rpe[m0 + c] ----
        v8f qrA = (v8f){}, qrB = (v8f){};
        {
            int mA = m0 + m16;            if (mA > 2 * NN - 2) mA = 2 * NN - 2;
            int mB = m0 + 16 + m16;       if (mB > 2 * NN - 2) mB = 2 * NN - 2;
            const __bf16* rA = rb + (size_t)mA * DD + 16 * hf;
            const __bf16* rB = rb + (size_t)mB * DD + 16 * hf;
            v16bf bA0 = *(const v16bf*)rA;
            v16bf bA1 = *(const v16bf*)(rA + 32);
            v16bf bB0 = *(const v16bf*)rB;
            v16bf bB1 = *(const v16bf*)(rB + 32);
            qrA = WMMA(aq0, bA0, qrA);  qrA = WMMA(aq1, bA1, qrA);
            qrB = WMMA(aq0, bB0, qrB);  qrB = WMMA(aq1, bB1, qrB);
            if (jt == 0) {  // only first iteration; afterwards qr_hi = previous qrA
                int mC = m0 + 32 + m16;  if (mC > 2 * NN - 2) mC = 2 * NN - 2;
                const __bf16* rC = rb + (size_t)mC * DD + 16 * hf;
                v16bf bC0 = *(const v16bf*)rC;
                v16bf bC1 = *(const v16bf*)(rC + 32);
                qr_hi = WMMA(aq0, bC0, qr_hi);
                qr_hi = WMMA(aq1, bC1, qr_hi);
            }
        }

        // park qr band in LDS (C-layout -> row-major gatherable)
        #pragma unroll
        for (int r = 0; r < 8; ++r) {
            int row = r + 8 * hf;
            lds_qr[wid][row][m16]      = qrA[r];
            lds_qr[wid][row][16 + m16] = qrB[r];
            lds_qr[wid][row][32 + m16] = qr_hi[r];
        }
        __builtin_amdgcn_wave_barrier();

#if HAVE_ASYNC_LDS
        // current buffer ready when only the 8 just-issued copies remain outstanding
        if (pre) { WAIT_ASYNC(8); } else { WAIT_ASYNC(0); }
        __builtin_amdgcn_wave_barrier();
#endif

        // ---- the two 16x16 score tiles of this pair ----
        #pragma unroll
        for (int p = 0; p < 2; ++p) {
            const int jtile = jt + p;
            const int j0 = jtile << 4;
            if (jtile <= it) {
#if HAVE_ASYNC_LDS
                const __bf16* krow = &lds_kst[wid][cur][(p * 16 + m16) * DD + 16 * hf];
#else
                const __bf16* krow = kb + hq + (size_t)(j0 + m16) * DD + 16 * hf;
#endif
                v16bf bk0 = *(const v16bf*)krow;
                v16bf bk1 = *(const v16bf*)(krow + 32);
                v8f s = (v8f){};
                s = WMMA(aq0, bk0, s);
                s = WMMA(aq1, bk1, s);

                const bool diag = (jtile == it);
                #pragma unroll
                for (int r = 0; r < 8; ++r) {
                    int row = r + 8 * hf;                    // di
                    // rpe gather: col = di - dj + 31 - 16*p  in [0,47)
                    float sv = s[r] + lds_qr[wid][row][row - m16 + 31 - 16 * p];
                    float wv = 1.0f + sv + 0.5f * sv * sv;   // Taylor-2 exp
                    if (diag && (j0 + m16 > i0 + row)) wv = 0.0f;  // causal
                    rs_part[r] += wv;
                    lds_w[wid][row][p * 16 + m16] = wv;
                }
            } else {  // p==1 past the diagonal: zero the upper half of W
                #pragma unroll
                for (int r = 0; r < 8; ++r)
                    lds_w[wid][r + 8 * hf][16 + m16] = 0.0f;
            }
        }
        __builtin_amdgcn_wave_barrier();

        // ---- O += W (16x32) x V (32x64), 4 N-tiles of 16 ----
        v16bf aw;
        #pragma unroll
        for (int e = 0; e < 16; ++e)
            aw[e] = tobf(lds_w[wid][m16][a_kmap(e, hf)]);

        const __bf16* vcol = vtb + (size_t)h * DD * NN + (size_t)m16 * NN
                             + j0pair + 16 * hf;
        #pragma unroll
        for (int t = 0; t < 4; ++t) {
            // lane's 16 K-elements (j = j0pair + 16hf + e) are contiguous in vtb
            v16bf bv = *(const v16bf*)(vcol + (size_t)(t * 16) * NN);
            acc_o[t] = WMMA(aw, bv, acc_o[t]);
        }

        qr_hi = qrA;  // next window's cols 32..47 == this window's cols 0..15
        cur ^= 1;
    }

    // ---- row-sum reduction across the 16 lanes of each half ----
    float rs[8];
    #pragma unroll
    for (int r = 0; r < 8; ++r) {
        float s = rs_part[r];
        s += __shfl_xor(s, 1, 32);
        s += __shfl_xor(s, 2, 32);
        s += __shfl_xor(s, 4, 32);
        s += __shfl_xor(s, 8, 32);
        rs[r] = s;
    }

    // ---- normalize + store ----
    float* obase = out + hq + (size_t)i0 * DD;
    #pragma unroll
    for (int t = 0; t < 4; ++t) {
        int d = t * 16 + m16;
        #pragma unroll
        for (int r = 0; r < 8; ++r) {
            int row = r + 8 * hf;
            obase[(size_t)row * DD + d] = acc_o[t][r] / rs[r];
        }
    }
}

extern "C" void kernel_launch(void* const* d_in, const int* in_sizes, int n_in,
                              void* d_out, int out_size, void* d_ws, size_t ws_size,
                              hipStream_t stream) {
    const float* q   = (const float*)d_in[0];
    const float* k   = (const float*)d_in[1];
    const float* v   = (const float*)d_in[2];
    const float* rpe = (const float*)d_in[3];
    // d_in[4] is the python scalar mask (==1, causal) — baked into the kernel.
    float* out = (float*)d_out;

    // workspace layout (bf16): kb [H*N*D] | vtb [H*D*N] | rb [(2N-1)*D]
    const int nkv = HH * NN * DD;            // 1,048,576
    const int nr  = (2 * NN - 1) * DD;       //   262,080
    __bf16* kb  = (__bf16*)d_ws;
    __bf16* vtb = kb + (size_t)nkv;
    __bf16* rbw = vtb + (size_t)nkv;

    cvt_bf16<<<dim3((nkv + 255) / 256), dim3(256), 0, stream>>>(k, kb, nkv);
    transpose_v_bf16<<<dim3((nkv + 255) / 256), dim3(256), 0, stream>>>(v, vtb);
    cvt_bf16<<<dim3((nr + 255) / 256), dim3(256), 0, stream>>>(rpe, rbw, nr);

    fastmax_fwd<<<dim3(HH * (NN / 64)), dim3(128), 0, stream>>>(q, kb, vtb, rbw, out);
}
